// Network_28114855919900
// MI455X (gfx1250) — matrix operand order repaired and checked by reading, not compiled, not run on hardware
//
#include <hip/hip_runtime.h>
#include <hip/hip_bf16.h>
#include <math.h>

// Problem constants (match reference)
#define N_  32768
#define P_  5
#define M_  1024
#define J_  24
#define H_  256
#define SMPL_THRESH 0.08f

typedef __attribute__((ext_vector_type(16))) __bf16 v16bf;
typedef __attribute__((ext_vector_type(8)))  __bf16 v8bf;
typedef __attribute__((ext_vector_type(8)))  float  v8f;

// ---------------------------------------------------------------------------
// Kernel W: one-time cast + transpose of W2 (f32, [k][n]) -> W2T (bf16, [n][k])
// 32x32 LDS tiles; 64 blocks x 256 threads. Negligible runtime, halves the
// per-block W2 traffic for the MLP and makes B fragments contiguous.
// ---------------------------------------------------------------------------
__global__ __launch_bounds__(256) void w2_cast_kernel(
    const float* __restrict__ W2, __bf16* __restrict__ W2T)
{
    __shared__ float tile[32][33];
    const int bt = blockIdx.x;          // 8x8 grid of 32x32 tiles
    const int kr = (bt >> 3) * 32;      // k base
    const int nc = (bt & 7) * 32;       // n base
    const int tx = threadIdx.x & 31;
    const int ty = threadIdx.x >> 5;    // 0..7
    #pragma unroll
    for (int i = 0; i < 32; i += 8)
        tile[ty + i][tx] = W2[(kr + ty + i) * H_ + nc + tx];
    __syncthreads();
    #pragma unroll
    for (int i = 0; i < 32; i += 8)
        W2T[(nc + ty + i) * H_ + kr + tx] = (__bf16)tile[tx][ty + i];
}

// ---------------------------------------------------------------------------
// Kernel A: nearest-neighbor search + LBS transforms.
// One thread per pose point n; part_pts staged per-part as padded float4 so
// the argmin loop uses ds_load_b128.
// ---------------------------------------------------------------------------
__global__ __launch_bounds__(256) void nn_lbs_kernel(
    const float* __restrict__ pose_pts,   // (N,3)
    const float* __restrict__ pose_dirs,  // (N,3)
    const float* __restrict__ part_pts,   // (P,M,3)
    const float* __restrict__ part_pbw,   // (P,M,J)
    const float* __restrict__ A,          // (J,4,4)
    const float* __restrict__ bigA,       // (J,4,4)
    const int*   __restrict__ lengths2,   // (P,)
    float* __restrict__ o_tdirs,          // (N*P,3)
    float* __restrict__ o_pflag,          // (N*P,)
    float* __restrict__ o_init,           // (N*P,3)
    float* __restrict__ o_pnorm)          // (N*P,)
{
    __shared__ __align__(16) float4 s_pts4[M_];   // 16 KB, one part at a time

    const int tid = threadIdx.x;
    const int n   = blockIdx.x * 256 + tid;       // grid is exact: n < N_

    const float x0 = pose_pts[n * 3 + 0];
    const float x1 = pose_pts[n * 3 + 1];
    const float x2 = pose_pts[n * 3 + 2];
    const float dx0 = pose_dirs[n * 3 + 0];
    const float dx1 = pose_dirs[n * 3 + 1];
    const float dx2 = pose_dirs[n * 3 + 2];

    for (int p = 0; p < P_; ++p) {
        // ---- stage this part's points, padded to float4 ----
        __syncthreads();
        for (int i = tid; i < M_; i += 256) {
            const float* q = &part_pts[(p * M_ + i) * 3];
            s_pts4[i] = make_float4(q[0], q[1], q[2], 0.0f);
        }
        __syncthreads();

        int len = lengths2[p];
        if (len > M_) len = M_;

        // ---- argmin over part points (b128 LDS broadcast reads) ----
        float best = 3.4e38f;
        int   bi = 0;
        #pragma unroll 4
        for (int m = 0; m < len; ++m) {
            const float4 y = s_pts4[m];
            const float a = x0 - y.x;
            const float b = x1 - y.y;
            const float c = x2 - y.z;
            const float d2 = fmaf(a, a, fmaf(b, b, c * c));
            if (d2 < best) { best = d2; bi = m; }
        }
        const float pnorm = sqrtf(best);
        const float pflag = (pnorm < SMPL_THRESH) ? 1.0f : 0.0f;

        // ---- blend transforms with bw = part_pbw[p, nn] ----
        const float* bwrow = &part_pbw[(p * M_ + bi) * J_];
        float Ab[12], Bb[12];
        #pragma unroll
        for (int e = 0; e < 12; ++e) { Ab[e] = 0.f; Bb[e] = 0.f; }
        for (int j = 0; j < J_; ++j) {
            const float w = bwrow[j];
            const float* Aj = &A[j * 16];
            const float* Gj = &bigA[j * 16];
            #pragma unroll
            for (int e = 0; e < 12; ++e) {
                Ab[e] = fmaf(w, Aj[e], Ab[e]);
                Bb[e] = fmaf(w, Gj[e], Bb[e]);
            }
        }

        // ---- invert 3x3 of Ab ----
        const float r00 = Ab[0], r01 = Ab[1], r02 = Ab[2],  t0 = Ab[3];
        const float r10 = Ab[4], r11 = Ab[5], r12 = Ab[6],  t1 = Ab[7];
        const float r20 = Ab[8], r21 = Ab[9], r22 = Ab[10], t2 = Ab[11];
        const float det = r00 * (r11 * r22 - r12 * r21)
                        - r01 * (r10 * r22 - r12 * r20)
                        + r02 * (r10 * r21 - r11 * r20);
        const float inv = 1.0f / det;
        const float i00 = (r11 * r22 - r12 * r21) * inv;
        const float i01 = (r02 * r21 - r01 * r22) * inv;
        const float i02 = (r01 * r12 - r02 * r11) * inv;
        const float i10 = (r12 * r20 - r10 * r22) * inv;
        const float i11 = (r00 * r22 - r02 * r20) * inv;
        const float i12 = (r02 * r10 - r00 * r12) * inv;
        const float i20 = (r10 * r21 - r11 * r20) * inv;
        const float i21 = (r01 * r20 - r00 * r21) * inv;
        const float i22 = (r00 * r11 - r01 * r10) * inv;

        // ---- tpts = Rinv @ (x - t) ; init = Rb @ tpts + tb ----
        const float px = x0 - t0, py = x1 - t1, pz = x2 - t2;
        const float tp0 = i00 * px + i01 * py + i02 * pz;
        const float tp1 = i10 * px + i11 * py + i12 * pz;
        const float tp2 = i20 * px + i21 * py + i22 * pz;
        const float init0 = Bb[0] * tp0 + Bb[1] * tp1 + Bb[2]  * tp2 + Bb[3];
        const float init1 = Bb[4] * tp0 + Bb[5] * tp1 + Bb[6]  * tp2 + Bb[7];
        const float init2 = Bb[8] * tp0 + Bb[9] * tp1 + Bb[10] * tp2 + Bb[11];

        // ---- dirs: Rb @ (Rinv @ d) ----
        const float td0 = i00 * dx0 + i01 * dx1 + i02 * dx2;
        const float td1 = i10 * dx0 + i11 * dx1 + i12 * dx2;
        const float td2 = i20 * dx0 + i21 * dx1 + i22 * dx2;
        const float od0 = Bb[0] * td0 + Bb[1] * td1 + Bb[2]  * td2;
        const float od1 = Bb[4] * td0 + Bb[5] * td1 + Bb[6]  * td2;
        const float od2 = Bb[8] * td0 + Bb[9] * td1 + Bb[10] * td2;

        const int row = n * P_ + p;
        o_tdirs[row * 3 + 0] = od0;
        o_tdirs[row * 3 + 1] = od1;
        o_tdirs[row * 3 + 2] = od2;
        o_init[row * 3 + 0] = init0;
        o_init[row * 3 + 1] = init1;
        o_init[row * 3 + 2] = init2;
        o_pflag[row] = pflag;
        o_pnorm[row] = pnorm;
    }
}

// ---------------------------------------------------------------------------
// Kernel B: fused MLP. 256 threads = 8 waves; 64 rows per block.
// Layer 2 (256x256) on v_wmma_f32_16x16x32_bf16. A fragments: b128 LDS loads
// (shared across a wave's 8 tiles). B fragments: contiguous b128 global loads
// from pre-transposed bf16 W2T (L2-resident) -- no staging, no barriers in
// the K loop, fully unrolled so loads pipeline across chunks.
// ---------------------------------------------------------------------------
#define NROWS  64
#define HPAD   264   // h pitch (bf16): 528 B/row, 16B aligned, staggers banks

__global__ __launch_bounds__(256) void mlp_kernel(
    const float*  __restrict__ W1, const float* __restrict__ b1,
    const __bf16* __restrict__ W2T,                  // (n,k) bf16
    const float*  __restrict__ b2,
    const float*  __restrict__ W3, const float* __restrict__ b3,
    const float*  __restrict__ d_init,   // (N*P,3)
    const float*  __restrict__ d_pflag,  // (N*P,)
    float* __restrict__ d_resd,          // (N*P,3)
    float* __restrict__ d_tpose)         // (N*P,3)
{
    __shared__ __align__(16) __bf16 sh_h[NROWS][HPAD];    // h1 then h2
    __shared__ __align__(16) float  sh_x[NROWS][3];
    __shared__ __align__(16) float  sh_w3[H_ * 3];

    const int tid  = threadIdx.x;
    const int row0 = blockIdx.x * NROWS;

    // ---- stage input rows + W3 ----
    for (int i = tid; i < NROWS * 3; i += 256) {
        sh_x[i / 3][i % 3] = d_init[(row0 + i / 3) * 3 + (i % 3)];
    }
    for (int i = tid; i < H_ * 3; i += 256) {
        sh_w3[i] = W3[i];
    }
    __syncthreads();

    // ---- layer 1: h1 = relu(x @ W1 + b1)
    // thread owns an 8-column group -> one b128 LDS store per row
    {
        const int cg = (tid & 31) * 8;     // column group base
        const int r0 = tid >> 5;           // 0..7
        float wv0[8], wv1[8], wv2[8], bbv[8];
        #pragma unroll
        for (int i = 0; i < 8; ++i) {
            wv0[i] = W1[cg + i];
            wv1[i] = W1[H_ + cg + i];
            wv2[i] = W1[2 * H_ + cg + i];
            bbv[i] = b1[cg + i];
        }
        #pragma unroll
        for (int rr = 0; rr < 8; ++rr) {
            const int r = r0 * 8 + rr;
            const float xx0 = sh_x[r][0], xx1 = sh_x[r][1], xx2 = sh_x[r][2];
            v8bf hv;
            #pragma unroll
            for (int i = 0; i < 8; ++i) {
                float acc = fmaf(xx2, wv2[i], fmaf(xx1, wv1[i], fmaf(xx0, wv0[i], bbv[i])));
                hv[i] = (__bf16)fmaxf(acc, 0.0f);
            }
            *(v8bf*)&sh_h[r][cg] = hv;
        }
    }
    __syncthreads();

    // ---- layer 2: h2 = relu(h1 @ W2 + b2) via WMMA bf16 ----
    const int wave = tid >> 5;
    const int lane = tid & 31;
    const int lm   = lane & 15;    // A: tile row; B/D: tile col
    const int lh   = lane >> 4;    // lane-half selector
    const int rt   = wave >> 1;    // row tile: constant per wave -> A reuse
    const int ctb  = (wave & 1) * 8;

    v8f acc[8];
    const v8f vzero = {0.f, 0.f, 0.f, 0.f, 0.f, 0.f, 0.f, 0.f};
    #pragma unroll
    for (int t = 0; t < 8; ++t) acc[t] = vzero;

    const int arow = rt * 16 + lm;

    #pragma unroll
    for (int kc = 0; kc < 8; ++kc) {           // K chunks of 32, fully unrolled
        // A fragment: shared by all 8 tiles of this wave (rt fixed).
        // v16bf halves contiguous: elems 0-7 -> k = lh*8.., 8-15 -> 16+lh*8..
        const v8bf alo = *(const v8bf*)&sh_h[arow][kc * 32 + lh * 8];
        const v8bf ahi = *(const v8bf*)&sh_h[arow][kc * 32 + 16 + lh * 8];
        const v16bf afrag = __builtin_shufflevector(
            alo, ahi, 0, 1, 2, 3, 4, 5, 6, 7, 8, 9, 10, 11, 12, 13, 14, 15);

        #pragma unroll
        for (int t = 0; t < 8; ++t) {
            const int ncol = (ctb + t) * 16 + lm;
            // B fragment: elems 0-7 -> k = kc*32 + lh*8.., 8-15 -> kc*32 + 16 + lh*8..
            const __bf16* bp = &W2T[ncol * H_ + kc * 32 + lh * 8];
            const v8bf blo = *(const v8bf*)bp;
            const v8bf bhi = *(const v8bf*)(bp + 16);
            const v16bf bfrag = __builtin_shufflevector(
                blo, bhi, 0, 1, 2, 3, 4, 5, 6, 7, 8, 9, 10, 11, 12, 13, 14, 15);
            acc[t] = __builtin_amdgcn_wmma_f32_16x16x32_bf16(
                false, afrag, false, bfrag, (short)0, acc[t], false, false);
        }
    }
    __syncthreads();   // all reads of sh_h(h1) complete before overwrite

    // ---- bias + relu, write h2 back into sh_h (bf16) ----
    #pragma unroll
    for (int t = 0; t < 8; ++t) {
        const int ncol = (ctb + t) * 16 + lm;
        const float bb2 = b2[ncol];
        #pragma unroll
        for (int v = 0; v < 8; ++v) {
            const int m = rt * 16 + lh * 8 + v;   // D layout: vgpr v, lane half
            sh_h[m][ncol] = (__bf16)fmaxf(acc[t][v] + bb2, 0.0f);
        }
    }
    __syncthreads();

    // ---- layer 3: resd = 0.05*tanh(h2 @ W3 + b3) * pflag ; tpose = init+resd
    {
        const int r = tid >> 2;
        const int q = tid & 3;
        if (q < 3) {
            float a3 = b3[q];
            for (int k8 = 0; k8 < H_ / 8; ++k8) {
                const v8bf hvv = *(const v8bf*)&sh_h[r][k8 * 8];
                #pragma unroll
                for (int i = 0; i < 8; ++i) {
                    a3 = fmaf((float)hvv[i], sh_w3[(k8 * 8 + i) * 3 + q], a3);
                }
            }
            const int grow = row0 + r;
            const float rd = 0.05f * tanhf(a3) * d_pflag[grow];
            d_resd[grow * 3 + q]  = rd;
            d_tpose[grow * 3 + q] = sh_x[r][q] + rd;
        }
    }
}

// ---------------------------------------------------------------------------
// Launch
// ---------------------------------------------------------------------------
extern "C" void kernel_launch(void* const* d_in, const int* in_sizes, int n_in,
                              void* d_out, int out_size, void* d_ws, size_t ws_size,
                              hipStream_t stream) {
    const float* pose_pts  = (const float*)d_in[0];
    const float* pose_dirs = (const float*)d_in[1];
    const float* part_pts  = (const float*)d_in[2];
    const float* part_pbw  = (const float*)d_in[3];
    const float* A         = (const float*)d_in[4];
    const float* bigA      = (const float*)d_in[5];
    const float* W1        = (const float*)d_in[6];
    const float* b1        = (const float*)d_in[7];
    const float* W2        = (const float*)d_in[8];
    const float* b2        = (const float*)d_in[9];
    const float* W3        = (const float*)d_in[10];
    const float* b3        = (const float*)d_in[11];
    const int*   lengths2  = (const int*)d_in[12];

    float* out = (float*)d_out;
    const int NP = N_ * P_;
    float* o_tpose = out;                      // (1,N,P,3)
    float* o_tdirs = out + (size_t)NP * 3;     // (1,N,P,3)
    float* o_resd  = out + (size_t)2 * NP * 3; // (1,N,P,3)
    float* o_pflag = out + (size_t)3 * NP * 3; // (1,N,P)
    float* o_init  = o_pflag + NP;             // (1,N*P,3)
    float* o_pnorm = o_init + (size_t)NP * 3;  // (1,N*P)

    __bf16* W2T = (__bf16*)d_ws;               // 128 KB scratch

    // Stage 0: one-time cast+transpose of W2 to bf16 (L2-resident afterwards)
    w2_cast_kernel<<<64, 256, 0, stream>>>(W2, W2T);

    // Stage 1: NN + LBS transforms (writes tdirs, pflag, init, pnorm)
    nn_lbs_kernel<<<N_ / 256, 256, 0, stream>>>(
        pose_pts, pose_dirs, part_pts, part_pbw, A, bigA, lengths2,
        o_tdirs, o_pflag, o_init, o_pnorm);

    // Stage 2: fused MLP with WMMA (writes resd, tpose)
    mlp_kernel<<<NP / NROWS, 256, 0, stream>>>(
        W1, b1, W2T, b2, W3, b3, o_init, o_pflag, o_resd, o_tpose);
}
